// DLRM_33998961115952
// MI455X (gfx1250) — compile-verified, matrix operand order checked by reading, not run on hardware
//
#include <hip/hip_runtime.h>
#include <cstdint>

#define B_BATCH 8192
#define NT 26
#define VOCAB 100000
#define D_EMB 128
#define P_POOL 10
#define NF 27
#define FEAT_STRIDE (NF * D_EMB) /* 3456 */
#define ZDIM 480                 /* 128 + 351, padded to 15*32 */

typedef __attribute__((ext_vector_type(16))) __bf16 v16bf;
typedef __attribute__((ext_vector_type(8)))  float  v8f;
typedef unsigned short ushort_t;
typedef int v4i __attribute__((vector_size(16)));
typedef __attribute__((address_space(1))) v4i* gp_v4i;
typedef __attribute__((address_space(3))) v4i* lp_v4i;

#if defined(__has_builtin)
#if __has_builtin(__builtin_amdgcn_global_load_async_to_lds_b128) && \
    __has_builtin(__builtin_amdgcn_s_wait_asynccnt)
#define USE_ASYNC_LDS 1
#endif
#endif

static __device__ __forceinline__ ushort_t f2bf(float f) {
  union { float f; unsigned u; } x; x.f = f;
  unsigned r = x.u + 0x7FFFu + ((x.u >> 16) & 1u); // round-to-nearest-even
  return (ushort_t)(r >> 16);
}
static __device__ __forceinline__ float bf2f(ushort_t h) {
  union { unsigned u; float f; } x; x.u = ((unsigned)h) << 16;
  return x.f;
}

// 16-byte global -> LDS copy: async (ASYNCcnt) on CDNA5, sync fallback.
// Generic->AS(1): global generic addresses are numerically identical.
// Generic->AS(3): LDS generic address carries the LDS offset in the low 32
// bits (ISA 10.2 aperture rules), so truncation yields the AS(3) offset.
static __device__ __forceinline__ void copy16_g2l(const ushort_t* g, ushort_t* l) {
#if defined(USE_ASYNC_LDS)
  __builtin_amdgcn_global_load_async_to_lds_b128(
      (gp_v4i)(uintptr_t)g, (lp_v4i)(unsigned int)(uintptr_t)l, 0, 0);
#else
  *(uint4*)l = *(const uint4*)g;
#endif
}
static __device__ __forceinline__ void wait_staged() {
#if defined(USE_ASYNC_LDS)
  __builtin_amdgcn_s_wait_asynccnt(0);
#endif
  __syncthreads();
}

union FragU { v16bf v; uint4 u[2]; };

// ISA 7.12.2 16-bit A/B fragment: lane L holds row (L&15); lanes 0-15 carry
// K = 0..7 and 16..23, lanes 16-31 carry K = 8..15 and 24..31 -> 2x b128 / lane.
static __device__ __forceinline__ v16bf load_frag(const ushort_t* base, int row0,
                                                  int k0, int ldk, int lane) {
  int r  = row0 + (lane & 15);
  int kb = k0 + ((lane >> 4) << 3);
  FragU f;
  f.u[0] = *(const uint4*)(base + r * ldk + kb);
  f.u[1] = *(const uint4*)(base + r * ldk + kb + 16);
  return f.v;
}

// ---------------- dense_x f32 (B x 13) -> bf16 (B x 32, zero padded) ----------
__global__ __launch_bounds__(256) void dlrm_pad_dense(const float* __restrict__ x,
                                                      ushort_t* __restrict__ xp) {
  int i = blockIdx.x * 256 + threadIdx.x;
  int r = i >> 5, c = i & 31;
  float v = (c < 13) ? x[r * 13 + c] : 0.f;
  xp[i] = f2bf(v);
}

// ---------------- weight prep: W (N x K f32) -> Wb (N x Kpad bf16, zero pad) --
__global__ __launch_bounds__(256) void dlrm_prep_w(const float* __restrict__ W,
                                                   ushort_t* __restrict__ Wb,
                                                   int K, int Kpad) {
  int i = blockIdx.x * 256 + threadIdx.x;
  int r = i / Kpad, c = i - r * Kpad;
  Wb[i] = (c < K) ? f2bf(W[(size_t)r * K + c]) : (ushort_t)0;
}

// ---------------- embedding bag pooling: one wave per (table, sample) --------
__global__ __launch_bounds__(256) void dlrm_embed(const int* __restrict__ sidx,
                                                  const float* __restrict__ tables,
                                                  ushort_t* __restrict__ feat) {
  int wave = blockIdx.x * 8 + (threadIdx.x >> 5);
  int lane = threadIdx.x & 31;
  int b = wave & (B_BATCH - 1);
  int t = wave >> 13;
  const int* ip = sidx + ((size_t)t * B_BATCH + b) * P_POOL;
  const float* tb = tables + (size_t)t * VOCAB * D_EMB;
  float a0 = 0.f, a1 = 0.f, a2 = 0.f, a3 = 0.f;
#pragma unroll
  for (int p = 0; p < P_POOL; ++p) {
    const float4* row = (const float4*)(tb + (size_t)ip[p] * D_EMB);
    float4 v = row[lane];
    a0 += v.x; a1 += v.y; a2 += v.z; a3 += v.w;
  }
  union { ushort_t s[4]; uint2 u; } o;
  o.s[0] = f2bf(a0); o.s[1] = f2bf(a1); o.s[2] = f2bf(a2); o.s[3] = f2bf(a3);
  *(uint2*)(feat + (size_t)b * FEAT_STRIDE + (1 + t) * D_EMB + lane * 4) = o.u;
}

// ---------------- WMMA GEMM: C = relu(A * Wb^T + bias), bf16 in/out ----------
// A:  M x kpad bf16 (padding pre-zeroed), Wb: N x kpad bf16 (padding pre-zeroed)
// Block: 256 thr (8 waves), tile 256x128; per-wave 64x64 (4x4 WMMA frags).
// Double-buffered LDS, async global->LDS staging, ONE barrier per k-tile:
// stage(kt+1) is issued after the barrier, which already ordered it against
// the last readers of its destination buffer (compute of tile kt-1).
__global__ __launch_bounds__(256) void dlrm_gemm_bf16(
    const ushort_t* __restrict__ A,
    const ushort_t* __restrict__ Wb,
    const float* __restrict__ bias,
    ushort_t* __restrict__ C, int ldc, int kpad) {
  __shared__ ushort_t sA[2][256 * 32]; // 16KB per buffer
  __shared__ ushort_t sW[2][128 * 32]; //  8KB per buffer
  int tid  = threadIdx.x;
  int lane = tid & 31;
  int wave = tid >> 5;
  int m_blk = blockIdx.x * 256;
  int n_blk = blockIdx.y * 128;
  int wm = (wave & 3) * 64;  // 4 waves tile M
  int wn = (wave >> 2) * 64; // 2 waves tile N

  v8f zero8 = {0.f, 0.f, 0.f, 0.f, 0.f, 0.f, 0.f, 0.f};
  v8f acc[4][4];
#pragma unroll
  for (int i = 0; i < 4; ++i)
#pragma unroll
    for (int j = 0; j < 4; ++j) acc[i][j] = zero8;

  int ktiles = kpad >> 5;
  int rS = tid >> 2;           // staging: 4 lanes cover one 64B row segment
  int cS = (tid & 3) * 8;      // element column within the 32-wide tile

  // stage one 32-wide k-tile into LDS buffer `buf` (A: 256 rows, W: 128 rows)
  auto stage = [&](int kt, int buf) {
    int k0 = kt * 32;
#pragma unroll
    for (int s = 0; s < 4; ++s) {
      int r = s * 64 + rS;
      copy16_g2l(A + (size_t)(m_blk + r) * kpad + k0 + cS, &sA[buf][r * 32 + cS]);
    }
#pragma unroll
    for (int s = 0; s < 2; ++s) {
      int r = s * 64 + rS;
      copy16_g2l(Wb + (size_t)(n_blk + r) * kpad + k0 + cS, &sW[buf][r * 32 + cS]);
    }
  };

  stage(0, 0);
  for (int kt = 0; kt < ktiles; ++kt) {
    wait_staged(); // asynccnt==0 + barrier: tile kt resident, all waves synced
    if (kt + 1 < ktiles) stage(kt + 1, (kt + 1) & 1); // streams under the WMMAs
    const ushort_t* bA = sA[kt & 1];
    const ushort_t* bW = sW[kt & 1];
    v16bf af[4], bf[4];
#pragma unroll
    for (int i = 0; i < 4; ++i) af[i] = load_frag(bA, wm + i * 16, 0, 32, lane);
#pragma unroll
    for (int j = 0; j < 4; ++j) bf[j] = load_frag(bW, wn + j * 16, 0, 32, lane);
#pragma unroll
    for (int i = 0; i < 4; ++i)
#pragma unroll
      for (int j = 0; j < 4; ++j)
        acc[i][j] = __builtin_amdgcn_wmma_f32_16x16x32_bf16(
            false, af[i], false, bf[j], (short)0, acc[i][j], false, false);
  }

  // epilogue: bias + relu + bf16 store (C/D layout: vgpr v -> row, lane&15 -> col)
  int col = lane & 15;
  int rh  = (lane >> 4) << 3;
#pragma unroll
  for (int tj = 0; tj < 4; ++tj) {
    int n = n_blk + wn + tj * 16 + col;
    float bv = bias[n];
#pragma unroll
    for (int ti = 0; ti < 4; ++ti) {
#pragma unroll
      for (int v = 0; v < 8; ++v) {
        int m = m_blk + wm + ti * 16 + rh + v;
        float xv = acc[ti][tj][v] + bv;
        xv = xv > 0.f ? xv : 0.f;
        C[(size_t)m * ldc + n] = f2bf(xv);
      }
    }
  }
}

// ---------------- pairwise interactions: Z = T T^T, upper triangle ----------
// One wave per sample; T is 27x128 (padded to 32x128 in LDS). Symmetric, so
// only tiles (0,0),(0,1),(1,1) are computed: 12 WMMAs per sample.
__global__ __launch_bounds__(128) void dlrm_interact(const ushort_t* __restrict__ feat,
                                                     ushort_t* __restrict__ z) {
  __shared__ ushort_t sT[4][32 * D_EMB];
  int lane = threadIdx.x & 31;
  int wave = threadIdx.x >> 5;
  int b = blockIdx.x * 4 + wave;
  ushort_t* S = sT[wave];
  { // stage: lane -> feature row `lane` (zeros for rows 27..31)
    uint4* dst = (uint4*)(S + lane * D_EMB);
    if (lane < NF) {
      const uint4* src = (const uint4*)(feat + (size_t)b * FEAT_STRIDE + lane * D_EMB);
#pragma unroll
      for (int i = 0; i < 16; ++i) dst[i] = src[i];
    } else {
      uint4 zv = {0u, 0u, 0u, 0u};
#pragma unroll
      for (int i = 0; i < 16; ++i) dst[i] = zv;
    }
  }
  __syncthreads();
  { // z[0:64] = first 64 elems of bottom-MLP output (feature row 0); zero the pad column 479
    ushort_t* zb = z + (size_t)b * ZDIM;
    if (lane < 8) ((uint4*)zb)[lane] = ((const uint4*)S)[lane];
    if (lane == 8) zb[ZDIM - 1] = 0;
  }
  v8f zero8 = {0.f, 0.f, 0.f, 0.f, 0.f, 0.f, 0.f, 0.f};
  v8f acc00 = zero8, acc01 = zero8, acc11 = zero8;
#pragma unroll
  for (int kt = 0; kt < 4; ++kt) {
    v16bf f0 = load_frag(S, 0,  kt * 32, D_EMB, lane);
    v16bf f1 = load_frag(S, 16, kt * 32, D_EMB, lane);
    acc00 = __builtin_amdgcn_wmma_f32_16x16x32_bf16(false, f0, false, f0, (short)0, acc00, false, false);
    acc01 = __builtin_amdgcn_wmma_f32_16x16x32_bf16(false, f0, false, f1, (short)0, acc01, false, false);
    acc11 = __builtin_amdgcn_wmma_f32_16x16x32_bf16(false, f1, false, f1, (short)0, acc11, false, false);
  }
  // scatter triu(27, k=1): off(m,n) = m*26 - m(m-1)/2 + n - m - 1
  ushort_t* zi = z + (size_t)b * ZDIM + D_EMB;
  int col = lane & 15;
  int rh  = (lane >> 4) << 3;
#pragma unroll
  for (int v = 0; v < 8; ++v) {
    int m = rh + v;
    { int n = col;                                   // tile (0,0): m,n in 0..15
      if (n > m) zi[m * 26 - (m * (m - 1)) / 2 + n - m - 1] = f2bf(acc00[v]); }
    { int n = 16 + col;                              // tile (0,1): always n > m
      if (n < NF) zi[m * 26 - (m * (m - 1)) / 2 + n - m - 1] = f2bf(acc01[v]); }
    { int mm = 16 + m; int n = 16 + col;             // tile (1,1)
      if (n > mm && n < NF) zi[mm * 26 - (mm * (mm - 1)) / 2 + n - mm - 1] = f2bf(acc11[v]); }
  }
}

// ---------------- final 256 -> 1 dot + sigmoid: one wave per sample ----------
__global__ __launch_bounds__(256) void dlrm_final(const ushort_t* __restrict__ h,
                                                  const float* __restrict__ w,
                                                  const float* __restrict__ bias,
                                                  float* __restrict__ out) {
  int wave = blockIdx.x * 8 + (threadIdx.x >> 5);
  int lane = threadIdx.x & 31;
  const ushort_t* hp = h + (size_t)wave * 256 + lane * 8;
  const float* wp = w + lane * 8;
  float s = 0.f;
#pragma unroll
  for (int j = 0; j < 8; ++j) s += bf2f(hp[j]) * wp[j];
#pragma unroll
  for (int off = 16; off > 0; off >>= 1) s += __shfl_xor(s, off, 32);
  if (lane == 0) out[wave] = 1.f / (1.f + expf(-(s + bias[0])));
}

extern "C" void kernel_launch(void* const* d_in, const int* in_sizes, int n_in,
                              void* d_out, int out_size, void* d_ws, size_t ws_size,
                              hipStream_t stream) {
  (void)in_sizes; (void)n_in; (void)out_size; (void)ws_size;
  const float* dense_x = (const float*)d_in[0];
  const int*   sidx    = (const int*)d_in[1];
  const float* emb     = (const float*)d_in[2];
  const float* bw0 = (const float*)d_in[3];  const float* bb0 = (const float*)d_in[4];
  const float* bw1 = (const float*)d_in[5];  const float* bb1 = (const float*)d_in[6];
  const float* bw2 = (const float*)d_in[7];  const float* bb2 = (const float*)d_in[8];
  const float* tw0 = (const float*)d_in[9];  const float* tb0 = (const float*)d_in[10];
  const float* tw1 = (const float*)d_in[11]; const float* tb1 = (const float*)d_in[12];
  const float* tw2 = (const float*)d_in[13]; const float* tb2 = (const float*)d_in[14];
  const float* tw3 = (const float*)d_in[15]; const float* tb3 = (const float*)d_in[16];
  const float* tw4 = (const float*)d_in[17]; const float* tb4 = (const float*)d_in[18];
  float* out = (float*)d_out;

  char* ws = (char*)d_ws;
  size_t off = 0;
  auto alloc = [&](size_t bytes) {
    char* p = ws + off;
    off = (off + bytes + 255) & ~(size_t)255;
    return p;
  };
  ushort_t* x0   = (ushort_t*)alloc((size_t)B_BATCH * 32 * 2);
  ushort_t* hb1  = (ushort_t*)alloc((size_t)B_BATCH * 512 * 2);
  ushort_t* hb2  = (ushort_t*)alloc((size_t)B_BATCH * 256 * 2);
  ushort_t* feat = (ushort_t*)alloc((size_t)B_BATCH * FEAT_STRIDE * 2);
  ushort_t* zb   = (ushort_t*)alloc((size_t)B_BATCH * ZDIM * 2);
  ushort_t* ht1  = (ushort_t*)alloc((size_t)B_BATCH * 1024 * 2);
  ushort_t* ht2  = (ushort_t*)alloc((size_t)B_BATCH * 1024 * 2);
  ushort_t* ht3  = (ushort_t*)alloc((size_t)B_BATCH * 512 * 2);
  ushort_t* ht4  = (ushort_t*)alloc((size_t)B_BATCH * 256 * 2);
  // bf16 padded weights
  ushort_t* wbB0 = (ushort_t*)alloc((size_t)512 * 32 * 2);
  ushort_t* wbB1 = (ushort_t*)alloc((size_t)256 * 512 * 2);
  ushort_t* wbB2 = (ushort_t*)alloc((size_t)128 * 256 * 2);
  ushort_t* wbT0 = (ushort_t*)alloc((size_t)1024 * 480 * 2);
  ushort_t* wbT1 = (ushort_t*)alloc((size_t)1024 * 1024 * 2);
  ushort_t* wbT2 = (ushort_t*)alloc((size_t)512 * 1024 * 2);
  ushort_t* wbT3 = (ushort_t*)alloc((size_t)256 * 512 * 2);

  // weight conversion (tiny; L2-resident afterwards)
  dlrm_prep_w<<<(512 * 32) / 256, 256, 0, stream>>>(bw0, wbB0, 13, 32);
  dlrm_prep_w<<<(256 * 512) / 256, 256, 0, stream>>>(bw1, wbB1, 512, 512);
  dlrm_prep_w<<<(128 * 256) / 256, 256, 0, stream>>>(bw2, wbB2, 256, 256);
  dlrm_prep_w<<<(1024 * 480) / 256, 256, 0, stream>>>(tw0, wbT0, 479, 480);
  dlrm_prep_w<<<(1024 * 1024) / 256, 256, 0, stream>>>(tw1, wbT1, 1024, 1024);
  dlrm_prep_w<<<(512 * 1024) / 256, 256, 0, stream>>>(tw2, wbT2, 1024, 1024);
  dlrm_prep_w<<<(256 * 512) / 256, 256, 0, stream>>>(tw3, wbT3, 512, 512);

  dlrm_pad_dense<<<(B_BATCH * 32) / 256, 256, 0, stream>>>(dense_x, x0);
  dlrm_embed<<<(NT * B_BATCH) / 8, 256, 0, stream>>>(sidx, emb, feat);
  // bottom MLP: 13(->32) -> 512 -> 256 -> 128 (last layer writes feature slot 0)
  dlrm_gemm_bf16<<<dim3(32, 4), 256, 0, stream>>>(x0,  wbB0, bb0, hb1,  512,         32);
  dlrm_gemm_bf16<<<dim3(32, 2), 256, 0, stream>>>(hb1, wbB1, bb1, hb2,  256,         512);
  dlrm_gemm_bf16<<<dim3(32, 1), 256, 0, stream>>>(hb2, wbB2, bb2, feat, FEAT_STRIDE, 256);
  dlrm_interact<<<B_BATCH / 4, 128, 0, stream>>>(feat, zb);
  // top MLP: 479(->480) -> 1024 -> 1024 -> 512 -> 256
  dlrm_gemm_bf16<<<dim3(32, 8), 256, 0, stream>>>(zb,  wbT0, tb0, ht1, 1024, 480);
  dlrm_gemm_bf16<<<dim3(32, 8), 256, 0, stream>>>(ht1, wbT1, tb1, ht2, 1024, 1024);
  dlrm_gemm_bf16<<<dim3(32, 4), 256, 0, stream>>>(ht2, wbT2, tb2, ht3, 512,  1024);
  dlrm_gemm_bf16<<<dim3(32, 2), 256, 0, stream>>>(ht3, wbT3, tb3, ht4, 256,  512);
  dlrm_final<<<B_BATCH / 8, 256, 0, stream>>>(ht4, tw4, tb4, out);
}
